// NeuralLongTermMemory_47321949667482
// MI455X (gfx1250) — matrix operand order repaired and checked by reading, not compiled, not run on hardware
//
#include <hip/hip_runtime.h>
#include <math.h>

// ---------------------------------------------------------------------------
// MI455X (gfx1250) NeuralLongTermMemory. All large GEMMs via
// V_WMMA_F32_16X16X32_BF16 (wave32, f32 accumulate). 128x128 block tiles,
// 8 waves x (32x64), BK=32. Double-buffered LDS fed by
// GLOBAL_LOAD_ASYNC_TO_LDS_B128 (ASYNCcnt-pipelined).
// ---------------------------------------------------------------------------

typedef __bf16 bf16;
typedef __bf16 v16bf __attribute__((ext_vector_type(16)));
typedef __bf16 v8bf  __attribute__((ext_vector_type(8)));
typedef float  v8f   __attribute__((ext_vector_type(8)));
typedef int    v4i   __attribute__((ext_vector_type(4)));

#define AS1 __attribute__((address_space(1)))
#define AS3 __attribute__((address_space(3)))

#if defined(__has_builtin)
#if __has_builtin(__builtin_amdgcn_global_load_async_to_lds_b128)
#define HAVE_ASYNC_COPY 1
#endif
#if __has_builtin(__builtin_amdgcn_s_wait_asynccnt)
#define HAVE_WAIT_ASYNC 1
#endif
#endif

#define BM 128
#define BN 128
#define BKD 32
#define PK 40   // padded LDS row stride in bf16 elems (80B, 16B aligned)

__device__ __forceinline__ float sigmoid_fast(float x) {
  return __builtin_amdgcn_rcpf(1.0f + __expf(-x));
}
__device__ __forceinline__ float silu_f(float x) { return x * sigmoid_fast(x); }

__device__ __forceinline__ void async_copy_b128(const bf16* g, bf16* l) {
#ifdef HAVE_ASYNC_COPY
  __builtin_amdgcn_global_load_async_to_lds_b128(
      (AS1 v4i*)(uintptr_t)g,
      (AS3 v4i*)(unsigned int)(uintptr_t)l, 0, 0);
#else
  *(v8bf*)l = *(const v8bf*)g;
#endif
}

template <int N>
__device__ __forceinline__ void async_wait_le() {
#ifdef HAVE_ASYNC_COPY
#ifdef HAVE_WAIT_ASYNC
  __builtin_amdgcn_s_wait_asynccnt(N);
#else
  asm volatile("s_wait_asynccnt %0" :: "i"(N) : "memory");
#endif
#endif
}

// ---------------------------------------------------------------------------
// bf16 WMMA GEMM:  C[m,n] = sum_k opA(m,k) * opB(n,k)
//   TA==0: opA(m,k) = A[m*lda + k]   (MxK row-major)   -> async double-buffer
//   TA==1: opA(m,k) = A[k*lda + m]   (KxM row-major)   -> sync transposed stage
// EPI: 0 = Cb raw bf16 ; 1 = Cb silu bf16 ; 2 = Cf raw f32 ;
//      3 = Cpre raw bf16 + Cb silu bf16.   C is MxN row-major (ldc).
// ---------------------------------------------------------------------------
template <int TA, int TB, int EPI>
__global__ __launch_bounds__(256)
void gemm_bf16_wmma(const bf16* __restrict__ A, const bf16* __restrict__ B,
                    int K, int lda, int ldb,
                    float* __restrict__ Cf, bf16* __restrict__ Cb,
                    bf16* __restrict__ Cpre, int ldc) {
  constexpr int NBUF = (TA == 0) ? 2 : 1;
  __shared__ bf16 sA[NBUF * BM * PK];
  __shared__ bf16 sB[NBUF * BN * PK];

  const int tid  = threadIdx.x;
  const int lane = tid & 31;
  const int wave = tid >> 5;
  const int wm   = wave >> 1;   // 0..3 wave row (32 rows)
  const int wn   = wave & 1;    // 0..1 wave col (64 cols)
  const int tileM = blockIdx.y * BM;
  const int tileN = blockIdx.x * BN;

  v8f acc[2][4] = {};

  // per-thread staging addresses, advanced incrementally per K-step
  const int m_c  = tid >> 2;          // mode0: rows m_c and m_c+64
  const int kc_c = (tid & 3) * 8;     // mode0: 8-elem k-chunk
  const int mt   = tid & 127;         // mode1: column within tile
  const int ktop = tid >> 7;          // mode1: k parity base

  const bf16* pA0 = nullptr; const bf16* pA1 = nullptr;
  bf16* qA0 = nullptr; bf16* qA1 = nullptr;
  const bf16* pB0 = nullptr; const bf16* pB1 = nullptr;
  bf16* qB0 = nullptr; bf16* qB1 = nullptr;

  if constexpr (TA == 0) {
    pA0 = A + (size_t)(tileM + m_c) * lda + kc_c;
    pA1 = A + (size_t)(tileM + 64 + m_c) * lda + kc_c;
    qA0 = &sA[m_c * PK + kc_c];
    qA1 = &sA[(64 + m_c) * PK + kc_c];
  } else {
    pA0 = A + (size_t)ktop * lda + tileM + mt;
    qA0 = &sA[mt * PK + ktop];
  }
  if constexpr (TB == 0) {
    pB0 = B + (size_t)(tileN + m_c) * ldb + kc_c;
    pB1 = B + (size_t)(tileN + 64 + m_c) * ldb + kc_c;
    qB0 = &sB[m_c * PK + kc_c];
    qB1 = &sB[(64 + m_c) * PK + kc_c];
  } else {
    pB0 = B + (size_t)ktop * ldb + tileN + mt;
    qB0 = &sB[mt * PK + ktop];
  }

  if constexpr (TA == 0) {
    // prologue: stage K-tile 0 into buffer 0
    async_copy_b128(pA0, qA0);
    async_copy_b128(pA1, qA1);
    async_copy_b128(pB0, qB0);
    async_copy_b128(pB1, qB1);
    pA0 += BKD; pA1 += BKD; pB0 += BKD; pB1 += BKD;
  }

  for (int k0 = 0; k0 < K; k0 += BKD) {
    int cur = 0;
    if constexpr (TA == 0) {
      cur = (k0 / BKD) & 1;
      const int nxt = cur ^ 1;
      if (k0 + BKD < K) {
        // issue next tile's copies, then wait only for the current tile's
        // 4 copies (async loads complete in order => <=4 outstanding means
        // the first group is done) while next-tile copies stay in flight.
        async_copy_b128(pA0, qA0 + nxt * (BM * PK));
        async_copy_b128(pA1, qA1 + nxt * (BM * PK));
        async_copy_b128(pB0, qB0 + nxt * (BN * PK));
        async_copy_b128(pB1, qB1 + nxt * (BN * PK));
        pA0 += BKD; pA1 += BKD; pB0 += BKD; pB1 += BKD;
        async_wait_le<4>();
      } else {
        async_wait_le<0>();
      }
    } else {
      // transposed staging: coalesced global reads along m, strided LDS write
      const bf16* pa = pA0;
      const bf16* pb = pB0;
      #pragma unroll
      for (int i = 0; i < 16; ++i) {
        qA0[2 * i] = pa[0];             // sA[mt][2i+ktop]
        qB0[2 * i] = pb[0];
        pa += 2 * (size_t)lda;
        pb += 2 * (size_t)ldb;
      }
      pA0 += (size_t)BKD * lda;
      pB0 += (size_t)BKD * ldb;
      __builtin_prefetch(pA0, 0, 1);    // next K tile (global_prefetch_b8)
      __builtin_prefetch(pB0, 0, 1);
    }

    __syncthreads();

    const bf16* bA = &sA[cur * (BM * PK)];
    const bf16* bB = &sB[cur * (BN * PK)];

    // ---- fragments per ISA 7.12.2, 8 WMMAs per K-step ----
    // A 16x32 bf16: lanes<16: elems0..7=K0..7, 8..15=K16..23
    //               lanes>=16: elems0..7=K8..15, 8..15=K24..31
    v16bf af[2];
    {
      const int kb = (lane < 16) ? 0 : 8;
      #pragma unroll
      for (int sm = 0; sm < 2; ++sm) {
        int r = wm * 32 + sm * 16 + (lane & 15);
        v8bf lo = *(const v8bf*)&bA[r * PK + kb];
        v8bf hi = *(const v8bf*)&bA[r * PK + 16 + kb];
        #pragma unroll
        for (int t = 0; t < 8; ++t) { af[sm][t] = lo[t]; af[sm][8 + t] = hi[t]; }
      }
    }
    // B 32x16 bf16: lanes<16: elems0..7=K0..7, 8..15=K16..23; lanes>=16: K8..15, K24..31
    v16bf bfrag[4];
    {
      const int kb2 = (lane < 16) ? 0 : 8;
      #pragma unroll
      for (int sn = 0; sn < 4; ++sn) {
        int c = wn * 64 + sn * 16 + (lane & 15);
        v8bf lo = *(const v8bf*)&bB[c * PK + kb2];
        v8bf hi = *(const v8bf*)&bB[c * PK + 16 + kb2];
        #pragma unroll
        for (int t = 0; t < 8; ++t) { bfrag[sn][t] = lo[t]; bfrag[sn][8 + t] = hi[t]; }
      }
    }
    #pragma unroll
    for (int sm = 0; sm < 2; ++sm)
      #pragma unroll
      for (int sn = 0; sn < 4; ++sn)
        acc[sm][sn] = __builtin_amdgcn_wmma_f32_16x16x32_bf16(
            false, af[sm], false, bfrag[sn], (short)0, acc[sm][sn], false, false);

    __syncthreads();
  }

  // ---- epilogue (fully specialized, no runtime branching) ----
  #pragma unroll
  for (int sm = 0; sm < 2; ++sm) {
    #pragma unroll
    for (int sn = 0; sn < 4; ++sn) {
      #pragma unroll
      for (int v = 0; v < 8; ++v) {
        int r = tileM + wm * 32 + sm * 16 + ((lane < 16) ? v : v + 8);
        int c = tileN + wn * 64 + sn * 16 + (lane & 15);
        float val = acc[sm][sn][v];
        size_t idx = (size_t)r * ldc + c;
        if constexpr (EPI == 0) {
          Cb[idx] = (bf16)val;
        } else if constexpr (EPI == 1) {
          Cb[idx] = (bf16)silu_f(val);
        } else if constexpr (EPI == 2) {
          Cf[idx] = val;
        } else {
          Cpre[idx] = (bf16)val;
          Cb[idx] = (bf16)silu_f(val);
        }
      }
    }
  }
}

// ---------------------------------------------------------------------------
// small helper kernels
// ---------------------------------------------------------------------------
__global__ void cvt_f32_bf16(const float* __restrict__ s, bf16* __restrict__ d, int n) {
  int i = blockIdx.x * 256 + threadIdx.x;
  if (i < n) d[i] = (bf16)s[i];
}

__global__ void transpose_cvt(const float* __restrict__ s, bf16* __restrict__ d,
                              int R, int C) {
  int i = blockIdx.x * 256 + threadIdx.x;
  if (i < R * C) {
    int r = i / C, c = i % C;
    d[(size_t)c * R + r] = (bf16)s[i];
  }
}

__global__ __launch_bounds__(256)
void l2norm_rows(bf16* __restrict__ x, int D) {
  __shared__ float red[256];
  bf16* p = x + (size_t)blockIdx.x * D;
  float s = 0.f;
  for (int i = threadIdx.x; i < D; i += 256) {
    float v = (float)p[i];
    s += v * v;
  }
  red[threadIdx.x] = s;
  __syncthreads();
  for (int off = 128; off > 0; off >>= 1) {
    if (threadIdx.x < off) red[threadIdx.x] += red[threadIdx.x + off];
    __syncthreads();
  }
  float inv = 1.f / fmaxf(sqrtf(red[0]), 1e-12f);
  for (int i = threadIdx.x; i < D; i += 256)
    p[i] = (bf16)((float)p[i] * inv);
}

__global__ void mean_over_s(const float* __restrict__ x, float* __restrict__ xm) {
  int idx = blockIdx.x * 256 + threadIdx.x;
  if (idx >= 4 * 1024) return;
  int b = idx >> 10, d = idx & 1023;
  const float* p = x + (size_t)b * 2048 * 1024 + d;
  float s = 0.f;
  for (int t = 0; t < 2048; ++t) s += p[(size_t)t * 1024];
  xm[idx] = s * (1.f / 2048.f);
}

__global__ __launch_bounds__(256)
void gates_kernel(const float* __restrict__ xm,
                  const float* __restrict__ wd, const float* __restrict__ bd,
                  const float* __restrict__ wl, const float* __restrict__ bl,
                  const float* __restrict__ wm, const float* __restrict__ bm,
                  float* __restrict__ out3) {
  __shared__ float red[256];
  const float* w; const float* b;
  if (blockIdx.x == 0)      { w = wd; b = bd; }
  else if (blockIdx.x == 1) { w = wl; b = bl; }
  else                      { w = wm; b = bm; }
  float acc = 0.f;
  for (int pair = threadIdx.x; pair < 4 * 1024; pair += 256) {
    int bb = pair >> 10, d = pair & 1023;
    const float* xr = xm + bb * 1024;
    const float* wr = w + (size_t)d * 1024;
    float dot = b[d];
    for (int k = 0; k < 1024; ++k) dot += xr[k] * wr[k];
    acc += sigmoid_fast(dot);
  }
  red[threadIdx.x] = acc;
  __syncthreads();
  for (int off = 128; off > 0; off >>= 1) {
    if (threadIdx.x < off) red[threadIdx.x] += red[threadIdx.x + off];
    __syncthreads();
  }
  if (threadIdx.x == 0) out3[blockIdx.x] = red[0] * (1.f / 4096.f);
}

__global__ void dpred_kernel(const bf16* __restrict__ pred, const bf16* __restrict__ values,
                             bf16* __restrict__ dpred, int n, float scale) {
  int i = blockIdx.x * 256 + threadIdx.x;
  if (i < n) dpred[i] = (bf16)(scale * ((float)pred[i] - (float)values[i]));
}

__global__ void silugrad_kernel(bf16* __restrict__ dh, const bf16* __restrict__ pre, int n) {
  int i = blockIdx.x * 256 + threadIdx.x;
  if (i < n) {
    float z = (float)pre[i];
    float s = sigmoid_fast(z);
    float g = s * (1.f + z * (1.f - s));
    dh[i] = (bf16)((float)dh[i] * g);
  }
}

__global__ void update_kernel(const float* __restrict__ w, const float* __restrict__ mom,
                              const float* __restrict__ g, const float* __restrict__ gates,
                              float* __restrict__ new_w, float* __restrict__ new_m, int n) {
  int i = blockIdx.x * 256 + threadIdx.x;
  if (i < n) {
    float alpha = gates[0], theta = gates[1], eta = gates[2];
    float nm = eta * mom[i] - theta * g[i];
    new_m[i] = nm;
    new_w[i] = (1.f - alpha) * w[i] + nm;
  }
}

// ---------------------------------------------------------------------------
extern "C" void kernel_launch(void* const* d_in, const int* in_sizes, int n_in,
                              void* d_out, int out_size, void* d_ws, size_t ws_size,
                              hipStream_t stream) {
  (void)in_sizes; (void)n_in; (void)out_size; (void)ws_size;
  const int MT = 4 * 2048;   // 8192 tokens
  const int D = 1024, H = 2048;

  const float* x     = (const float*)d_in[0];
  const float* kp1   = (const float*)d_in[1];
  const float* kp2   = (const float*)d_in[2];
  const float* vp1   = (const float*)d_in[3];
  const float* vp2   = (const float*)d_in[4];
  const float* qp1   = (const float*)d_in[5];
  const float* qp2   = (const float*)d_in[6];
  const float* gd_w  = (const float*)d_in[7];
  const float* gd_b  = (const float*)d_in[8];
  const float* gl_w  = (const float*)d_in[9];
  const float* gl_b  = (const float*)d_in[10];
  const float* gm_w  = (const float*)d_in[11];
  const float* gm_b  = (const float*)d_in[12];
  const float* memw0 = (const float*)d_in[13];
  const float* memw1 = (const float*)d_in[14];
  const float* mom0  = (const float*)d_in[15];
  const float* mom1  = (const float*)d_in[16];
  const float* outw  = (const float*)d_in[17];

  float* out    = (float*)d_out;
  float* new_w0 = out + (size_t)MT * D;
  float* new_w1 = new_w0 + (size_t)H * D;
  float* new_m0 = new_w1 + (size_t)D * H;
  float* new_m1 = new_m0 + (size_t)H * D;

  char* ws = (char*)d_ws;
  size_t off = 0;
  auto alloc = [&](size_t bytes) -> char* {
    char* p = ws + off;
    off += (bytes + 255) & ~(size_t)255;
    return p;
  };
  bf16* xb   = (bf16*)alloc((size_t)MT * D * 2);
  bf16* wk1  = (bf16*)alloc((size_t)D * D * 2);
  bf16* wk2  = (bf16*)alloc((size_t)D * D * 2);
  bf16* wv1  = (bf16*)alloc((size_t)D * D * 2);
  bf16* wv2  = (bf16*)alloc((size_t)D * D * 2);
  bf16* wq1  = (bf16*)alloc((size_t)D * D * 2);
  bf16* wq2  = (bf16*)alloc((size_t)D * D * 2);
  bf16* w0b  = (bf16*)alloc((size_t)H * D * 2);   // mem_w0 (H,D)
  bf16* w1b  = (bf16*)alloc((size_t)D * H * 2);   // mem_w1 (D,H)
  bf16* w1t  = (bf16*)alloc((size_t)H * D * 2);   // mem_w1^T (H,D)
  bf16* wob  = (bf16*)alloc((size_t)D * D * 2);
  bf16* tmp  = (bf16*)alloc((size_t)MT * D * 2);
  bf16* keys = (bf16*)alloc((size_t)MT * D * 2);
  bf16* vals = (bf16*)alloc((size_t)MT * D * 2);
  bf16* qrys = (bf16*)alloc((size_t)MT * D * 2);
  bf16* hq   = (bf16*)alloc((size_t)MT * H * 2);
  bf16* retr = (bf16*)alloc((size_t)MT * D * 2);
  bf16* prek = (bf16*)alloc((size_t)MT * H * 2);
  bf16* hk   = (bf16*)alloc((size_t)MT * H * 2);
  bf16* dh   = (bf16*)alloc((size_t)MT * H * 2);
  float* g0  = (float*)alloc((size_t)H * D * 4);
  float* g1  = (float*)alloc((size_t)D * H * 4);
  float* xm  = (float*)alloc(4 * 1024 * 4);
  float* gts = (float*)alloc(3 * 4);
  bf16* pred  = tmp;   // tmp free after projections
  bf16* dpred = retr;  // retr free after output projection

  auto cvt = [&](const float* s, bf16* d, int n) {
    cvt_f32_bf16<<<(n + 255) / 256, 256, 0, stream>>>(s, d, n);
  };
  // epi: 0 = Cb raw ; 1 = Cb silu ; 2 = Cf raw ; 3 = Cpre raw + Cb silu
  auto gemm = [&](const bf16* A, const bf16* Bm, int M_, int N_, int K_,
                  int lda, int ldb, int tA, int epi,
                  float* Cf, bf16* Cb, bf16* Cpre) {
    dim3 grid(N_ / BN, M_ / BM);
    if (tA) {
      gemm_bf16_wmma<1, 1, 2><<<grid, 256, 0, stream>>>(A, Bm, K_, lda, ldb,
                                                        Cf, Cb, Cpre, N_);
    } else {
      switch (epi) {
        case 0: gemm_bf16_wmma<0, 0, 0><<<grid, 256, 0, stream>>>(A, Bm, K_, lda, ldb, Cf, Cb, Cpre, N_); break;
        case 1: gemm_bf16_wmma<0, 0, 1><<<grid, 256, 0, stream>>>(A, Bm, K_, lda, ldb, Cf, Cb, Cpre, N_); break;
        case 2: gemm_bf16_wmma<0, 0, 2><<<grid, 256, 0, stream>>>(A, Bm, K_, lda, ldb, Cf, Cb, Cpre, N_); break;
        default: gemm_bf16_wmma<0, 0, 3><<<grid, 256, 0, stream>>>(A, Bm, K_, lda, ldb, Cf, Cb, Cpre, N_); break;
      }
    }
  };

  // ---- convert operands to bf16 ----
  cvt(x, xb, MT * D);
  cvt(kp1, wk1, D * D); cvt(kp2, wk2, D * D);
  cvt(vp1, wv1, D * D); cvt(vp2, wv2, D * D);
  cvt(qp1, wq1, D * D); cvt(qp2, wq2, D * D);
  cvt(memw0, w0b, H * D);
  cvt(memw1, w1b, D * H);
  cvt(outw, wob, D * D);
  transpose_cvt<<<(D * H + 255) / 256, 256, 0, stream>>>(memw1, w1t, D, H);

  // ---- K/V/Q projections: silu(silu(x@W1^T)@W2^T) ----
  gemm(xb, wk1, MT, D, D, D, D, 0, 1, nullptr, tmp, nullptr);
  gemm(tmp, wk2, MT, D, D, D, D, 0, 1, nullptr, keys, nullptr);
  gemm(xb, wv1, MT, D, D, D, D, 0, 1, nullptr, tmp, nullptr);
  gemm(tmp, wv2, MT, D, D, D, D, 0, 1, nullptr, vals, nullptr);
  gemm(xb, wq1, MT, D, D, D, D, 0, 1, nullptr, tmp, nullptr);
  gemm(tmp, wq2, MT, D, D, D, D, 0, 1, nullptr, qrys, nullptr);

  // ---- L2 normalize keys & queries ----
  l2norm_rows<<<MT, 256, 0, stream>>>(keys, D);
  l2norm_rows<<<MT, 256, 0, stream>>>(qrys, D);

  // ---- memory read path + output projection ----
  gemm(qrys, w0b, MT, H, D, D, D, 0, 1, nullptr, hq, nullptr);
  gemm(hq, w1b, MT, D, H, H, H, 0, 0, nullptr, retr, nullptr);
  gemm(retr, wob, MT, D, D, D, D, 0, 2, out, nullptr, nullptr);

  // ---- memory write path (grad of associative MSE loss) ----
  gemm(keys, w0b, MT, H, D, D, D, 0, 3, nullptr, hk, prek);
  gemm(hk, w1b, MT, D, H, H, H, 0, 0, nullptr, pred, nullptr);
  dpred_kernel<<<(MT * D + 255) / 256, 256, 0, stream>>>(
      pred, vals, dpred, MT * D, 2.f / (float)((size_t)MT * D));
  gemm(dpred, hk, D, H, MT, D, H, 1, 2, g1, nullptr, nullptr);
  gemm(dpred, w1t, MT, H, D, D, D, 0, 0, nullptr, dh, nullptr);
  silugrad_kernel<<<(MT * H + 255) / 256, 256, 0, stream>>>(dh, prek, MT * H);
  gemm(dh, keys, H, D, MT, H, D, 1, 2, g0, nullptr, nullptr);

  // ---- data-dependent gates ----
  mean_over_s<<<(4 * 1024 + 255) / 256, 256, 0, stream>>>(x, xm);
  gates_kernel<<<3, 256, 0, stream>>>(xm, gd_w, gd_b, gl_w, gl_b, gm_w, gm_b, gts);

  // ---- momentum + decayed weight updates ----
  update_kernel<<<(H * D + 255) / 256, 256, 0, stream>>>(
      memw0, mom0, g0, gts, new_w0, new_m0, H * D);
  update_kernel<<<(D * H + 255) / 256, 256, 0, stream>>>(
      memw1, mom1, g1, gts, new_w1, new_m1, D * H);
}
